// LSTM_23304492548205
// MI455X (gfx1250) — matrix operand order repaired and checked by reading, compiled, not run on hardware
//
#include <hip/hip_runtime.h>
#include <math.h>

#define Hsz 512
#define Bsz 64
#define Tsz 512
#define D0  128
#define OUTD 128

typedef __attribute__((ext_vector_type(16))) __bf16 v16bf;
typedef __attribute__((ext_vector_type(8)))  __bf16 v8bf;
typedef __attribute__((ext_vector_type(8)))  float  v8f;

__device__ __forceinline__ float sigmoidf_(float x) {
    return 1.0f / (1.0f + __expf(-x));
}

__device__ __forceinline__ v8f wmma_bf16(v16bf a, v16bf b, v8f c) {
    return __builtin_amdgcn_wmma_f32_16x16x32_bf16(false, a, false, b,
                                                   (short)0, c, false, false);
}

// A-operand loader: 16x32 bf16 A-matrix, contiguous K order matching load_b16.
// lanes 0-15 hold K=[kk..kk+15]; lanes 16-31 hold K=[kk+16..kk+31].
__device__ __forceinline__ v16bf load_a16(const __bf16* __restrict__ row, int kk, int kh) {
    return *reinterpret_cast<const v16bf*>(row + kk + kh * 16);
}

// B-operand loader: 32x16 bf16 B-matrix; lane (n = lane&15) holds its output
// column's contiguous K run (weights stored [N][K]); works for global or LDS.
__device__ __forceinline__ v16bf load_b16(const __bf16* row, int kk, int kh) {
    return *reinterpret_cast<const v16bf*>(row + kk + kh * 16);
}

// ---------------- conversion / setup kernels ----------------

__global__ void cvt_bf16(const float* __restrict__ in, __bf16* __restrict__ out, int n) {
    int i = blockIdx.x * blockDim.x + threadIdx.x;
    int stride = gridDim.x * blockDim.x;
    for (; i < n; i += stride) out[i] = (__bf16)in[i];
}

__global__ void bias_sum(const float* __restrict__ a, const float* __restrict__ b,
                         float* __restrict__ out, int n) {
    int i = blockIdx.x * blockDim.x + threadIdx.x;
    if (i < n) out[i] = a[i] + b[i];
}

__global__ void init_state(__bf16* __restrict__ h0, float* __restrict__ c, int n,
                           unsigned* __restrict__ ctr) {
    int i = blockIdx.x * blockDim.x + threadIdx.x;
    if (i < n) { h0[i] = (__bf16)0.0f; c[i] = 0.0f; }
    if (blockIdx.x == 0 && threadIdx.x == 0) *ctr = 0u;
}

// ---------------- persistent LSTM layer ----------------
// One block per 16-wide h-column tile (grid = 32), 256 threads = 8 wave32.
// The block's 64 weight rows (4 gates x 16 cols) of both Wih and Whh are staged
// into LDS once and stay resident for all T steps. Wave w owns M-tile (w&3)
// and gates (w>>2) and (w>>2)+2. Each gate uses TWO independent accumulator
// chains (even/odd 32-wide K chunks) so dependent WMMAs are >=3 instructions
// apart, covering the bf16 WMMA hazard window without v_nop filler.
// Timesteps are separated by a device-scope release/acquire barrier.
__global__ void lstm_layer_persistent(
        const __bf16* __restrict__ xseq, int xRowStride, int Din,
        const __bf16* __restrict__ Wih, const __bf16* __restrict__ Whh,
        const float*  __restrict__ bsum,
        __bf16* __restrict__ hb0, __bf16* __restrict__ hb1,
        float*  __restrict__ c,
        __bf16* __restrict__ hseq, int hseqRowStride,   // may be null
        unsigned* __restrict__ ctr, int T) {
    extern __shared__ char smem[];
    __bf16* whh_s = (__bf16*)smem;                       // 64 x Hsz
    __bf16* wih_s = whh_s + 64 * Hsz;                    // 64 x Din
    float*  gt    = (float*)(wih_s + (size_t)64 * Din);  // [4 mtile][4 gate][16][16]

    const int n0  = blockIdx.x * 16;
    const int tid = threadIdx.x;

    // ---- stage weight slices into LDS (once per layer) ----
    for (int cidx = tid; cidx < 64 * (Hsz / 8); cidx += blockDim.x) {
        int s = cidx >> 6;                // Hsz/8 = 64 16B-chunks per row
        int o = cidx & 63;
        int grow = (s >> 4) * Hsz + n0 + (s & 15);
        ((v8bf*)whh_s)[cidx] = ((const v8bf*)(Whh + (size_t)grow * Hsz))[o];
    }
    const int dchunks = Din >> 3;
    for (int cidx = tid; cidx < 64 * dchunks; cidx += blockDim.x) {
        int s = cidx / dchunks;
        int o = cidx - s * dchunks;
        int grow = (s >> 4) * Hsz + n0 + (s & 15);
        ((v8bf*)wih_s)[cidx] = ((const v8bf*)(Wih + (size_t)grow * Din))[o];
    }
    __syncthreads();

    const int wv = tid >> 5, lane = tid & 31;
    const int mtile = wv & 3, g0 = wv >> 2;   // gates g0 and g0+2
    const int r = lane & 15, kh = lane >> 4;
    const int m0 = mtile * 16;

    const __bf16* b0 = wih_s + (size_t)(g0 * 16 + r) * Din;
    const __bf16* b1 = wih_s + (size_t)((g0 + 2) * 16 + r) * Din;
    const __bf16* w0 = whh_s + (size_t)(g0 * 16 + r) * Hsz;
    const __bf16* w1 = whh_s + (size_t)((g0 + 2) * 16 + r) * Hsz;

    for (int t = 0; t < T; ++t) {
        const __bf16* hp = (t & 1) ? hb1 : hb0;
        __bf16*       hn = (t & 1) ? hb0 : hb1;

        v8f acc0e = {}, acc1e = {}, acc0o = {}, acc1o = {};

        // input contribution: x_t @ Wih^T  (A from global, B from LDS)
        const __bf16* arow = xseq + (size_t)t * Din + (size_t)(m0 + r) * xRowStride;
        #pragma unroll 2
        for (int kk = 0; kk < Din; kk += 64) {
            v16bf av0 = load_a16(arow, kk, kh);
            v16bf av1 = load_a16(arow, kk + 32, kh);
            acc0e = wmma_bf16(av0, load_b16(b0, kk, kh),      acc0e);
            acc1e = wmma_bf16(av0, load_b16(b1, kk, kh),      acc1e);
            acc0o = wmma_bf16(av1, load_b16(b0, kk + 32, kh), acc0o);
            acc1o = wmma_bf16(av1, load_b16(b1, kk + 32, kh), acc1o);
        }
        // recurrent contribution: h_{t-1} @ Whh^T
        const __bf16* hrow = hp + (size_t)(m0 + r) * Hsz;
        #pragma unroll 4
        for (int kk = 0; kk < Hsz; kk += 64) {
            __builtin_prefetch(hrow + kk + 128, 0, 0);
            v16bf av0 = load_a16(hrow, kk, kh);
            v16bf av1 = load_a16(hrow, kk + 32, kh);
            acc0e = wmma_bf16(av0, load_b16(w0, kk, kh),      acc0e);
            acc1e = wmma_bf16(av0, load_b16(w1, kk, kh),      acc1e);
            acc0o = wmma_bf16(av1, load_b16(w0, kk + 32, kh), acc0o);
            acc1o = wmma_bf16(av1, load_b16(w1, kk + 32, kh), acc1o);
        }

        // C/D layout: element (M = v + 8*kh, N = lane&15) lives in VGPR v
        #pragma unroll
        for (int v = 0; v < 8; ++v) {
            gt[(mtile * 4 + g0)     * 256 + (v + 8 * kh) * 16 + r] = acc0e[v] + acc0o[v];
            gt[(mtile * 4 + g0 + 2) * 256 + (v + 8 * kh) * 16 + r] = acc1e[v] + acc1o[v];
        }
        __syncthreads();

        // fused pointwise LSTM update: 1024 elements, 256 threads -> 4 each
        for (int e = tid; e < 1024; e += 256) {
            const int mt = e >> 8, mm = (e >> 4) & 15, nn = e & 15;
            const int hcol = n0 + nn;
            const int brow = mt * 16 + mm;
            float ig = sigmoidf_(gt[(mt * 4 + 0) * 256 + mm * 16 + nn] + bsum[hcol]);
            float fg = sigmoidf_(gt[(mt * 4 + 1) * 256 + mm * 16 + nn] + bsum[Hsz + hcol]);
            float gg = tanhf    (gt[(mt * 4 + 2) * 256 + mm * 16 + nn] + bsum[2 * Hsz + hcol]);
            float og = sigmoidf_(gt[(mt * 4 + 3) * 256 + mm * 16 + nn] + bsum[3 * Hsz + hcol]);
            const size_t idx = (size_t)brow * Hsz + hcol;
            float cn = fg * c[idx] + ig * gg;
            c[idx] = cn;
            float hv = og * tanhf(cn);
            hn[idx] = (__bf16)hv;
            if (hseq)
                hseq[(size_t)brow * hseqRowStride + (size_t)t * Hsz + hcol] = (__bf16)hv;
        }

        // ---- device-wide barrier between timesteps ----
        __threadfence();
        __syncthreads();
        if (tid == 0) {
            __hip_atomic_fetch_add(ctr, 1u, __ATOMIC_RELEASE, __HIP_MEMORY_SCOPE_AGENT);
            const unsigned tgt = (unsigned)(t + 1) * gridDim.x;
            while (__hip_atomic_load(ctr, __ATOMIC_ACQUIRE, __HIP_MEMORY_SCOPE_AGENT) < tgt)
                __builtin_amdgcn_s_sleep(1);
        }
        __syncthreads();
    }
}

// ---------------- output projection on last timestep ----------------
__global__ void out_gemm(const __bf16* __restrict__ hlast,
                         const __bf16* __restrict__ Wout,
                         const float*  __restrict__ bout,
                         float* __restrict__ out) {
    const int m0   = blockIdx.x * 16;   // 0..3
    const int n0   = blockIdx.y * 16;   // 0..7
    const int lane = threadIdx.x & 31;
    const int r    = lane & 15;
    const int kh   = lane >> 4;

    v8f acce = {}, acco = {};
    const __bf16* arow = hlast + (size_t)(m0 + r) * Hsz;
    const __bf16* brow = Wout + (size_t)(n0 + r) * Hsz;
    #pragma unroll 4
    for (int kk = 0; kk < Hsz; kk += 64) {
        acce = wmma_bf16(load_a16(arow, kk, kh),      load_b16(brow, kk, kh),      acce);
        acco = wmma_bf16(load_a16(arow, kk + 32, kh), load_b16(brow, kk + 32, kh), acco);
    }
    #pragma unroll
    for (int v = 0; v < 8; ++v) {
        const int mm = v + 8 * kh;
        out[(size_t)(m0 + mm) * OUTD + n0 + r] = acce[v] + acco[v] + bout[n0 + r];
    }
}

// ---------------- host launcher ----------------

extern "C" void kernel_launch(void* const* d_in, const int* in_sizes, int n_in,
                              void* d_out, int out_size, void* d_ws, size_t ws_size,
                              hipStream_t stream) {
    (void)in_sizes; (void)n_in; (void)out_size; (void)ws_size;

    const float* x    = (const float*)d_in[0];
    const float* Wih0 = (const float*)d_in[1];
    const float* Whh0 = (const float*)d_in[2];
    const float* bih0 = (const float*)d_in[3];
    const float* bhh0 = (const float*)d_in[4];
    const float* Wih1 = (const float*)d_in[5];
    const float* Whh1 = (const float*)d_in[6];
    const float* bih1 = (const float*)d_in[7];
    const float* bhh1 = (const float*)d_in[8];
    const float* Wout = (const float*)d_in[9];
    const float* bout = (const float*)d_in[10];
    float* out = (float*)d_out;

    char* ws = (char*)d_ws;
    size_t off = 0;
    auto alloc = [&](size_t bytes) {
        char* p = ws + off;
        off = (off + bytes + 255) & ~(size_t)255;
        return p;
    };

    __bf16*   Wih0b = (__bf16*)alloc((size_t)4 * Hsz * D0  * 2);
    __bf16*   Whh0b = (__bf16*)alloc((size_t)4 * Hsz * Hsz * 2);
    __bf16*   Wih1b = (__bf16*)alloc((size_t)4 * Hsz * Hsz * 2);
    __bf16*   Whh1b = (__bf16*)alloc((size_t)4 * Hsz * Hsz * 2);
    __bf16*   Woutb = (__bf16*)alloc((size_t)OUTD * Hsz * 2);
    __bf16*   xb    = (__bf16*)alloc((size_t)Bsz * Tsz * D0  * 2);
    __bf16*   hseq  = (__bf16*)alloc((size_t)Bsz * Tsz * Hsz * 2);
    __bf16*   hb0   = (__bf16*)alloc((size_t)Bsz * Hsz * 2);
    __bf16*   hb1   = (__bf16*)alloc((size_t)Bsz * Hsz * 2);
    float*    cbuf  = (float*) alloc((size_t)Bsz * Hsz * 4);
    float*    bsum0 = (float*) alloc((size_t)4 * Hsz * 4);
    float*    bsum1 = (float*) alloc((size_t)4 * Hsz * 4);
    unsigned* ctr0  = (unsigned*)alloc(256);
    unsigned* ctr1  = (unsigned*)alloc(256);

    // weight / input conversion to bf16
    cvt_bf16<<<512,  256, 0, stream>>>(Wih0, Wih0b, 4 * Hsz * D0);
    cvt_bf16<<<2048, 256, 0, stream>>>(Whh0, Whh0b, 4 * Hsz * Hsz);
    cvt_bf16<<<2048, 256, 0, stream>>>(Wih1, Wih1b, 4 * Hsz * Hsz);
    cvt_bf16<<<2048, 256, 0, stream>>>(Whh1, Whh1b, 4 * Hsz * Hsz);
    cvt_bf16<<<256,  256, 0, stream>>>(Wout, Woutb, OUTD * Hsz);
    cvt_bf16<<<2048, 256, 0, stream>>>(x,    xb,    Bsz * Tsz * D0);
    bias_sum<<<8, 256, 0, stream>>>(bih0, bhh0, bsum0, 4 * Hsz);
    bias_sum<<<8, 256, 0, stream>>>(bih1, bhh1, bsum1, 4 * Hsz);

    const size_t smem0 = (size_t)64 * Hsz * 2 + (size_t)64 * D0  * 2 + 4 * 4 * 256 * 4;
    const size_t smem1 = (size_t)64 * Hsz * 2 + (size_t)64 * Hsz * 2 + 4 * 4 * 256 * 4;

    // ---- layer 0 ----
    init_state<<<128, 256, 0, stream>>>(hb0, cbuf, Bsz * Hsz, ctr0);
    lstm_layer_persistent<<<32, 256, smem0, stream>>>(
        xb, Tsz * D0, D0, Wih0b, Whh0b, bsum0,
        hb0, hb1, cbuf, hseq, Tsz * Hsz, ctr0, Tsz);

    // ---- layer 1 ----
    init_state<<<128, 256, 0, stream>>>(hb0, cbuf, Bsz * Hsz, ctr1);
    lstm_layer_persistent<<<32, 256, smem1, stream>>>(
        hseq, Tsz * Hsz, Hsz, Wih1b, Whh1b, bsum1,
        hb0, hb1, cbuf, (__bf16*)nullptr, 0, ctr1, Tsz);

    // T=512 even -> final h lives in hb0
    out_gemm<<<dim3(4, 8), 32, 0, stream>>>(hb0, Woutb, bout, out);
}
